// Model_87385404604620
// MI455X (gfx1250) — matrix operand order repaired and checked by reading, not compiled, not run on hardware
//
#include <hip/hip_runtime.h>
#include <hip/hip_bf16.h>
#include <math.h>

// ---------------------------------------------------------------------------
// Types for WMMA
// ---------------------------------------------------------------------------
typedef __attribute__((ext_vector_type(16))) __bf16        v16bf;
typedef __attribute__((ext_vector_type(8)))  float         v8f;
typedef __attribute__((ext_vector_type(8)))  unsigned int  v8u;
typedef __attribute__((ext_vector_type(4)))  unsigned int  v4u;

__device__ __forceinline__ unsigned short f2bf(float f) {
  unsigned int u = __builtin_bit_cast(unsigned int, f);
  unsigned int r = u + 0x7FFFu + ((u >> 16) & 1u);   // round-to-nearest-even
  return (unsigned short)(r >> 16);
}
__device__ __forceinline__ unsigned int pack2(float a, float b) {
  return (unsigned int)f2bf(a) | ((unsigned int)f2bf(b) << 16);
}
// branchless guarded load (clamped index + select; no exec-mask branches)
__device__ __forceinline__ float ldg_guard(const float* __restrict__ p, long long idx, bool ok) {
  float v = p[ok ? idx : 0];
  return ok ? v : 0.0f;
}

// ---------------------------------------------------------------------------
// Batched GEMM:  D = alpha * A(MxK) @ B^T + bias + residual
//   A row-major [M,K] (lda).  B row-major [N,K] (ldb)  (always N-major).
//   Batch z: b1 = z/nb2, b2 = z%nb2; operand offsets b1*s?1 + b2*s?2.
// Block tile 64x128, BK=32, 256 threads = 8 waves, each wave 32x32 out
// -> 4 x v_wmma_f32_16x16x32_bf16 per wave per K-step.
// ---------------------------------------------------------------------------
#define BM 64
#define BN 128
#define BK 32

__global__ __launch_bounds__(256) void gemm_bf16_wmma(
    const float* __restrict__ A, const float* __restrict__ Bm,
    float* __restrict__ D, const float* __restrict__ bias,
    const float* __restrict__ Rsd,
    int M, int N, int Kd,
    int lda, int ldb, int ldd, int ldr,
    long long sA1, long long sA2, long long sB1, long long sB2,
    long long sD1, long long sD2, long long sR1, long long sR2,
    int nb2, float alpha)
{
  __shared__ unsigned short sA[BM * BK];   // [m][k]
  __shared__ unsigned short sB[BN * BK];   // [n][k]

  const int bz = blockIdx.z;
  const int b1 = bz / nb2, b2 = bz % nb2;
  const float* Ab = A  + b1 * sA1 + b2 * sA2;
  const float* Bb = Bm + b1 * sB1 + b2 * sB2;
  float*       Db = D  + b1 * sD1 + b2 * sD2;
  const float* Rb = Rsd ? (Rsd + b1 * sR1 + b2 * sR2) : (const float*)0;

  const int mBase = blockIdx.y * BM;
  const int nBase = blockIdx.x * BN;
  const int tid  = threadIdx.x;
  const int wid  = tid >> 5;
  const int lane = tid & 31;
  const int half = lane >> 4;
  const int r16  = lane & 15;
  const int mw   = wid & 1;        // 0..1 : 32-row slab
  const int nw   = wid >> 1;       // 0..3 : 32-col slab
  const int arow0 = mw * 32 + r16, arow1 = arow0 + 16;
  const int brow0 = nw * 32 + r16, brow1 = brow0 + 16;
  const int kbA  = half * 8;       // A frag K: {kbA..kbA+7} U {16+kbA..}
  const int kbB  = half * 8;       // B frag K: {kbB..kbB+7} U {16+kbB..}

  // staging coords
  const int aRow = tid >> 2, aCb = (tid & 3) * 8;     // A: 8 elems/thread
  const int bRow = tid >> 1, bCb = (tid & 1) * 16;    // B: 16 elems/thread

  v8f acc00 = {}, acc01 = {}, acc10 = {}, acc11 = {};
  const bool interior = (mBase + BM <= M) && (nBase + BN <= N);

  for (int k0 = 0; k0 < Kd; k0 += BK) {
    if (interior && (k0 + BK <= Kd)) {
      { // A tile: two b128 loads -> one b128 LDS store
        const float* p = Ab + (long long)(mBase + aRow) * lda + k0 + aCb;
        float4 x0 = *(const float4*)p;
        float4 x1 = *(const float4*)(p + 4);
        v4u u;
        u[0] = pack2(x0.x, x0.y); u[1] = pack2(x0.z, x0.w);
        u[2] = pack2(x1.x, x1.y); u[3] = pack2(x1.z, x1.w);
        *(v4u*)&sA[aRow * BK + aCb] = u;
      }
      { // B tile: four b128 loads -> two b128 LDS stores
        const float* p = Bb + (long long)(nBase + bRow) * ldb + k0 + bCb;
        float4 x0 = *(const float4*)p;
        float4 x1 = *(const float4*)(p + 4);
        float4 x2 = *(const float4*)(p + 8);
        float4 x3 = *(const float4*)(p + 12);
        v4u u0, u1;
        u0[0] = pack2(x0.x, x0.y); u0[1] = pack2(x0.z, x0.w);
        u0[2] = pack2(x1.x, x1.y); u0[3] = pack2(x1.z, x1.w);
        u1[0] = pack2(x2.x, x2.y); u1[1] = pack2(x2.z, x2.w);
        u1[2] = pack2(x3.x, x3.y); u1[3] = pack2(x3.z, x3.w);
        *(v4u*)&sB[bRow * BK + bCb]     = u0;
        *(v4u*)&sB[bRow * BK + bCb + 8] = u1;
      }
    } else {
      { // guarded A (branchless)
        long long gr = (long long)mBase + aRow;
        bool rok = gr < M;
        v4u u;
        #pragma unroll
        for (int j = 0; j < 4; ++j) {
          int kk = k0 + aCb + 2 * j;
          float v0 = ldg_guard(Ab, gr * lda + kk,     rok && (kk     < Kd));
          float v1 = ldg_guard(Ab, gr * lda + kk + 1, rok && (kk + 1 < Kd));
          u[j] = pack2(v0, v1);
        }
        *(v4u*)&sA[aRow * BK + aCb] = u;
      }
      { // guarded B (branchless)
        long long gn = (long long)nBase + bRow;
        bool nok = gn < N;
        v4u u0, u1;
        #pragma unroll
        for (int j = 0; j < 4; ++j) {
          int kk = k0 + bCb + 2 * j;
          float v0 = ldg_guard(Bb, gn * ldb + kk,     nok && (kk     < Kd));
          float v1 = ldg_guard(Bb, gn * ldb + kk + 1, nok && (kk + 1 < Kd));
          u0[j] = pack2(v0, v1);
        }
        #pragma unroll
        for (int j = 0; j < 4; ++j) {
          int kk = k0 + bCb + 8 + 2 * j;
          float v0 = ldg_guard(Bb, gn * ldb + kk,     nok && (kk     < Kd));
          float v1 = ldg_guard(Bb, gn * ldb + kk + 1, nok && (kk + 1 < Kd));
          u1[j] = pack2(v0, v1);
        }
        *(v4u*)&sB[bRow * BK + bCb]     = u0;
        *(v4u*)&sB[bRow * BK + bCb + 8] = u1;
      }
    }
    __syncthreads();

    // fragments (aligned v4u -> ds_load_b128)
    v4u a0l = *(const v4u*)&sA[arow0 * BK + kbA];
    v4u a0h = *(const v4u*)&sA[arow0 * BK + 16 + kbA];
    v4u a1l = *(const v4u*)&sA[arow1 * BK + kbA];
    v4u a1h = *(const v4u*)&sA[arow1 * BK + 16 + kbA];
    v4u b0l = *(const v4u*)&sB[brow0 * BK + kbB];
    v4u b0h = *(const v4u*)&sB[brow0 * BK + 16 + kbB];
    v4u b1l = *(const v4u*)&sB[brow1 * BK + kbB];
    v4u b1h = *(const v4u*)&sB[brow1 * BK + 16 + kbB];
    v8u au0 = {a0l[0],a0l[1],a0l[2],a0l[3],a0h[0],a0h[1],a0h[2],a0h[3]};
    v8u au1 = {a1l[0],a1l[1],a1l[2],a1l[3],a1h[0],a1h[1],a1h[2],a1h[3]};
    v8u bu0 = {b0l[0],b0l[1],b0l[2],b0l[3],b0h[0],b0h[1],b0h[2],b0h[3]};
    v8u bu1 = {b1l[0],b1l[1],b1l[2],b1l[3],b1h[0],b1h[1],b1h[2],b1h[3]};
    v16bf a0 = __builtin_bit_cast(v16bf, au0);
    v16bf a1 = __builtin_bit_cast(v16bf, au1);
    v16bf b0 = __builtin_bit_cast(v16bf, bu0);
    v16bf b1 = __builtin_bit_cast(v16bf, bu1);

    acc00 = __builtin_amdgcn_wmma_f32_16x16x32_bf16(false, a0, false, b0, (short)0, acc00, false, false);
    acc01 = __builtin_amdgcn_wmma_f32_16x16x32_bf16(false, a0, false, b1, (short)0, acc01, false, false);
    acc10 = __builtin_amdgcn_wmma_f32_16x16x32_bf16(false, a1, false, b0, (short)0, acc10, false, false);
    acc11 = __builtin_amdgcn_wmma_f32_16x16x32_bf16(false, a1, false, b1, (short)0, acc11, false, false);
    __syncthreads();
  }

  // epilogue: lanes 0-15 -> M=r, lanes 16-31 -> M=r+8 inside each 16-tile
  const int rb0 = mBase + mw * 32;
  const int c0  = nBase + nw * 32 + r16;
  const int c1  = c0 + 16;
  if (interior) {
    #pragma unroll
    for (int rr = 0; rr < 8; ++rr) {
      long long r0 = rb0 + (half ? rr + 8 : rr);
      long long r1 = r0 + 16;
      float v00 = alpha * acc00[rr], v01 = alpha * acc01[rr];
      float v10 = alpha * acc10[rr], v11 = alpha * acc11[rr];
      if (bias) { v00 += bias[c0]; v01 += bias[c1]; v10 += bias[c0]; v11 += bias[c1]; }
      if (Rb) {
        v00 += Rb[r0 * ldr + c0]; v01 += Rb[r0 * ldr + c1];
        v10 += Rb[r1 * ldr + c0]; v11 += Rb[r1 * ldr + c1];
      }
      Db[r0 * ldd + c0] = v00; Db[r0 * ldd + c1] = v01;
      Db[r1 * ldd + c0] = v10; Db[r1 * ldd + c1] = v11;
    }
  } else {
    #pragma unroll
    for (int rr = 0; rr < 8; ++rr) {
      long long r0 = rb0 + (half ? rr + 8 : rr);
      long long r1 = r0 + 16;
      float vv[4] = { acc00[rr], acc01[rr], acc10[rr], acc11[rr] };
      long long rws[4] = { r0, r0, r1, r1 };
      int cs[4] = { c0, c1, c0, c1 };
      #pragma unroll
      for (int q = 0; q < 4; ++q) {
        if (rws[q] < M && cs[q] < N) {
          float v = alpha * vv[q];
          if (bias) v += bias[cs[q]];
          if (Rb)   v += Rb[rws[q] * ldr + cs[q]];
          Db[rws[q] * ldd + cs[q]] = v;
        }
      }
    }
  }
}

// ---------------------------------------------------------------------------
// weight transpose: src [K,N] -> dst [N,K]
// ---------------------------------------------------------------------------
__global__ void transpose_kernel(const float* __restrict__ src, float* __restrict__ dst,
                                 int K, int N, long long total)
{
  long long i = (long long)blockIdx.x * 256 + threadIdx.x;
  if (i >= total) return;
  int k = (int)(i % K);
  long long n = i / K;
  dst[i] = src[(long long)k * N + n];
}

// ---------------------------------------------------------------------------
// LayerNorm: one 256-thread block per row
// ---------------------------------------------------------------------------
__global__ __launch_bounds__(256) void ln_kernel(
    const float* __restrict__ src, float* __restrict__ dst,
    const float* __restrict__ g, const float* __restrict__ b, int width)
{
  long long row = blockIdx.x;
  const float* x = src + row * width;
  float*       y = dst + row * width;
  float s = 0.f, s2 = 0.f;
  for (int i = threadIdx.x; i < width; i += 256) { float v = x[i]; s += v; s2 += v * v; }
  #pragma unroll
  for (int o = 16; o > 0; o >>= 1) { s += __shfl_xor(s, o, 32); s2 += __shfl_xor(s2, o, 32); }
  __shared__ float ps[8], ps2[8];
  int wid = threadIdx.x >> 5, lane = threadIdx.x & 31;
  if (lane == 0) { ps[wid] = s; ps2[wid] = s2; }
  __syncthreads();
  float ts = 0.f, ts2 = 0.f;
  #pragma unroll
  for (int i = 0; i < 8; ++i) { ts += ps[i]; ts2 += ps2[i]; }
  float mean = ts / width;
  float var  = ts2 / width - mean * mean;
  float rstd = rsqrtf(var + 1e-5f);
  for (int i = threadIdx.x; i < width; i += 256) {
    float v = (x[i] - mean) * rstd;
    y[i] = v * g[i] + b[i];
  }
}

// ---------------------------------------------------------------------------
// In-place row softmax
// ---------------------------------------------------------------------------
__global__ __launch_bounds__(256) void softmax_kernel(float* __restrict__ buf, int L)
{
  long long row = blockIdx.x;
  float* x = buf + row * L;
  float m = -1e30f;
  for (int i = threadIdx.x; i < L; i += 256) m = fmaxf(m, x[i]);
  #pragma unroll
  for (int o = 16; o > 0; o >>= 1) m = fmaxf(m, __shfl_xor(m, o, 32));
  __shared__ float pm[8], psum[8];
  int wid = threadIdx.x >> 5, lane = threadIdx.x & 31;
  if (lane == 0) pm[wid] = m;
  __syncthreads();
  float mm = pm[0];
  #pragma unroll
  for (int i = 1; i < 8; ++i) mm = fmaxf(mm, pm[i]);
  float s = 0.f;
  for (int i = threadIdx.x; i < L; i += 256) { float e = __expf(x[i] - mm); x[i] = e; s += e; }
  #pragma unroll
  for (int o = 16; o > 0; o >>= 1) s += __shfl_xor(s, o, 32);
  if (lane == 0) psum[wid] = s;
  __syncthreads();
  float ts = 0.f;
  #pragma unroll
  for (int i = 0; i < 8; ++i) ts += psum[i];
  float inv = 1.f / ts;
  for (int i = threadIdx.x; i < L; i += 256) x[i] *= inv;
}

// ---------------------------------------------------------------------------
// tmhsa group gather: src [10,2304,512] -> dst [256(g,h),400,128]
// padded rows (hh>=36) were zero before the projection -> value == bias[c]
// ---------------------------------------------------------------------------
__global__ void tmhsa_gather(const float* __restrict__ src,
                             const float* __restrict__ bias,
                             float* __restrict__ dst)
{
  long long i = (long long)blockIdx.x * 256 + threadIdx.x;
  const long long total = 256LL * 400 * 128;
  if (i >= total) return;
  int d = (int)(i & 127);
  long long t1 = i >> 7;
  int l  = (int)(t1 % 400);
  int gb = (int)(t1 / 400);
  int h  = gb & 3, g = gb >> 2;
  int g1 = g >> 3, g2 = g & 7;
  int t   = l / 40, rem = l % 40;
  int whi = rem >> 3, wwi = rem & 7;
  int hh = g1 * 5 + whi, w = g2 * 8 + wwi;
  int c = h * 128 + d;
  dst[i] = (hh < 36) ? src[((long long)t * 2304 + hh * 64 + w) * 512 + c] : bias[c];
}

// transposed variant for V: dst [256(g,h)][128][400]
__global__ void tmhsa_gather_t(const float* __restrict__ src,
                               const float* __restrict__ bias,
                               float* __restrict__ dst)
{
  long long i = (long long)blockIdx.x * 256 + threadIdx.x;
  const long long total = 256LL * 400 * 128;
  if (i >= total) return;
  int l = (int)(i % 400);
  long long t1 = i / 400;
  int d  = (int)(t1 % 128);
  int gb = (int)(t1 / 128);
  int h  = gb & 3, g = gb >> 2;
  int g1 = g >> 3, g2 = g & 7;
  int t   = l / 40, rem = l % 40;
  int whi = rem >> 3, wwi = rem & 7;
  int hh = g1 * 5 + whi, w = g2 * 8 + wwi;
  int c = h * 128 + d;
  dst[i] = (hh < 36) ? src[((long long)t * 2304 + hh * 64 + w) * 512 + c] : bias[c];
}

__global__ void tmhsa_scatter(const float* __restrict__ og, float* __restrict__ dst)
{
  long long i = (long long)blockIdx.x * 256 + threadIdx.x;
  const long long total = 23040LL * 512;
  if (i >= total) return;
  int c = (int)(i & 511);
  long long t1 = i >> 9;
  int n = (int)(t1 % 2304);
  int t = (int)(t1 / 2304);
  int hh = n >> 6, w = n & 63;
  int h = c >> 7, d = c & 127;
  int g1 = hh / 5, whi = hh % 5;
  int g2 = w >> 3, wwi = w & 7;
  int g = g1 * 8 + g2;
  int l = t * 40 + whi * 8 + wwi;
  dst[i] = og[(((long long)(g * 4 + h)) * 400 + l) * 128 + d];
}

// ---------------------------------------------------------------------------
// fold + normalization: y1 [10,2304,1960] -> fn [10,40,108,192]
// ---------------------------------------------------------------------------
__global__ void foldnorm_kernel(const float* __restrict__ y1, float* __restrict__ fn)
{
  long long i = (long long)blockIdx.x * 256 + threadIdx.x;
  const long long total = 10LL * 40 * 108 * 192;
  if (i >= total) return;
  int ow = (int)(i % 192);
  long long t1 = i / 192;
  int oh = (int)(t1 % 108); t1 /= 108;
  int cc = (int)(t1 % 40);
  int bt = (int)(t1 / 40);
  int pr = oh + 3, pc = ow + 3;
  int kis[3], kjs[3], nr = 0, nc = 0;
  for (int ki = pr % 3; ki < 7; ki += 3) {
    if (ki <= pr) { int gh = (pr - ki) / 3; if (gh < 36) kis[nr++] = ki; }
  }
  for (int kj = pc % 3; kj < 7; kj += 3) {
    if (kj <= pc) { int gw = (pc - kj) / 3; if (gw < 64) kjs[nc++] = kj; }
  }
  float sum = 0.f;
  for (int a = 0; a < nr; ++a)
    for (int b = 0; b < nc; ++b) {
      int ki = kis[a], kj = kjs[b];
      int gh = (pr - ki) / 3, gw = (pc - kj) / 3;
      sum += y1[((long long)bt * 2304 + gh * 64 + gw) * 1960 + cc * 49 + ki * 7 + kj];
    }
  fn[i] = sum / (float)(nr * nc);
}

// unfold + exact GeLU: fn [10,40,108,192] -> y2 [10,2304,1960]
__global__ void unfold_gelu_kernel(const float* __restrict__ fn, float* __restrict__ y2)
{
  long long i = (long long)blockIdx.x * 256 + threadIdx.x;
  const long long total = 23040LL * 1960;
  if (i >= total) return;
  int hd = (int)(i % 1960);
  long long t1 = i / 1960;
  int n  = (int)(t1 % 2304);
  int bt = (int)(t1 / 2304);
  int gh = n >> 6, gw = n & 63;
  int cc = hd / 49, r49 = hd % 49;
  int ki = r49 / 7, kj = r49 % 7;
  int rr = 3 * gh + ki - 3, c = 3 * gw + kj - 3;
  float v = 0.f;
  if (rr >= 0 && rr < 108 && c >= 0 && c < 192)
    v = fn[(((long long)bt * 40 + cc) * 108 + rr) * 192 + c];
  y2[i] = 0.5f * v * (1.0f + erff(v * 0.70710678118654752f));
}

// ---------------------------------------------------------------------------
// swmhsa helpers
// ---------------------------------------------------------------------------
__global__ void concat_xf_kernel(const float* __restrict__ x, const float* __restrict__ f,
                                 float* __restrict__ xf)
{
  long long i = (long long)blockIdx.x * 256 + threadIdx.x;
  const long long total = 23040LL * 768;
  if (i >= total) return;
  int c = (int)(i % 768);
  long long row = i / 768;
  xf[i] = (c < 512) ? x[row * 512 + c] : f[row * 256 + (c - 512)];
}

__global__ void gate_apply_kernel(const float* __restrict__ f, const float* __restrict__ gate,
                                  float* __restrict__ xf)
{
  long long i = (long long)blockIdx.x * 256 + threadIdx.x;
  const long long total = 23040LL * 256;
  if (i >= total) return;
  int j = (int)(i % 256);
  long long row = i / 256;
  float gv = gate[i];
  xf[row * 768 + 512 + j] = f[i] * (1.f / (1.f + __expf(-gv)));
}

// 4x4 depthwise conv, stride 4, src NHWC [10,36,64,srcLd] -> dst [10,144,cn]
__global__ void dwconv_kernel(const float* __restrict__ src, int srcLd,
                              const float* __restrict__ w, const float* __restrict__ b,
                              float* __restrict__ dst, int cn)
{
  long long i = (long long)blockIdx.x * 256 + threadIdx.x;
  const long long total = 10LL * 144 * cn;
  if (i >= total) return;
  int ch = (int)(i % cn);
  long long t1 = i / cn;
  int pix = (int)(t1 % 144);
  int bt  = (int)(t1 / 144);
  int oy = pix / 16, ox = pix % 16;
  float s = b[ch];
  #pragma unroll
  for (int ii = 0; ii < 4; ++ii)
    #pragma unroll
    for (int jj = 0; jj < 4; ++jj)
      s += src[((long long)bt * 2304 + (4 * oy + ii) * 64 + 4 * ox + jj) * srcLd + ch]
           * w[ch * 16 + ii * 4 + jj];
  dst[i] = s;
}

// window gather: src [10,2304,srcLd] -> dst [10,40,64,cn] (zero pad h>=36)
__global__ void win_gather_kernel(const float* __restrict__ src, int srcLd, int cn,
                                  float* __restrict__ dst, long long total)
{
  long long i = (long long)blockIdx.x * 256 + threadIdx.x;
  if (i >= total) return;
  int c = (int)(i % cn);
  long long t1 = i / cn;
  int p   = (int)(t1 % 64); t1 /= 64;
  int win = (int)(t1 % 40);
  int bt  = (int)(t1 / 40);
  int wy = win >> 3, wx = win & 7;
  int r = p >> 3, cp = p & 7;
  int h = wy * 8 + r, w = wx * 8 + cp;
  dst[i] = (h < 36) ? src[((long long)bt * 2304 + h * 64 + w) * srcLd + c] : 0.f;
}

// K concat: [10,40,208,512]; rows 0..63 = windowed, 64..207 = global tokens
__global__ void kvcat_kernel(const float* __restrict__ px, const float* __restrict__ pg,
                             float* __restrict__ dst)
{
  long long i = (long long)blockIdx.x * 256 + threadIdx.x;
  const long long total = 10LL * 40 * 208 * 512;
  if (i >= total) return;
  int c = (int)(i & 511);
  long long t1 = i >> 9;
  int rr  = (int)(t1 % 208); t1 /= 208;
  int win = (int)(t1 % 40);
  int bt  = (int)(t1 / 40);
  dst[i] = (rr < 64) ? px[(((long long)bt * 40 + win) * 64 + rr) * 512 + c]
                     : pg[((long long)bt * 144 + (rr - 64)) * 512 + c];
}

// V concat, transposed: dst [10,40,512,208]
__global__ void kvcat_t_kernel(const float* __restrict__ px, const float* __restrict__ pg,
                               float* __restrict__ dst)
{
  long long i = (long long)blockIdx.x * 256 + threadIdx.x;
  const long long total = 10LL * 40 * 512 * 208;
  if (i >= total) return;
  int rr = (int)(i % 208);
  long long t1 = i / 208;
  int c   = (int)(t1 % 512); t1 /= 512;
  int win = (int)(t1 % 40);
  int bt  = (int)(t1 / 40);
  dst[i] = (rr < 64) ? px[(((long long)bt * 40 + win) * 64 + rr) * 512 + c]
                     : pg[((long long)bt * 144 + (rr - 64)) * 512 + c];
}

__global__ void swin_scatter(const float* __restrict__ og, float* __restrict__ dst)
{
  long long i = (long long)blockIdx.x * 256 + threadIdx.x;
  const long long total = 23040LL * 512;
  if (i >= total) return;
  int c = (int)(i & 511);
  long long t1 = i >> 9;
  int n  = (int)(t1 % 2304);
  int bt = (int)(t1 / 2304);
  int hh = n >> 6, w = n & 63;
  int wy = hh >> 3, r = hh & 7, wx = w >> 3, cp = w & 7;
  int win = wy * 8 + wx, p = r * 8 + cp;
  dst[i] = og[(((long long)bt * 40 + win) * 64 + p) * 512 + c];
}

// ---------------------------------------------------------------------------
// Host side
// ---------------------------------------------------------------------------
static void gemm(hipStream_t st, const float* A, const float* B, float* D,
                 const float* bias, const float* R,
                 int M, int N, int K, int lda, int ldb, int ldd, int ldr,
                 int batch, int nb2,
                 long long sA1, long long sA2, long long sB1, long long sB2,
                 long long sD1, long long sD2, long long sR1, long long sR2,
                 float alpha)
{
  dim3 g((N + BN - 1) / BN, (M + BM - 1) / BM, batch);
  gemm_bf16_wmma<<<g, 256, 0, st>>>(A, B, D, bias, R, M, N, K, lda, ldb, ldd, ldr,
                                    sA1, sA2, sB1, sB2, sD1, sD2, sR1, sR2,
                                    nb2, alpha);
}

static inline unsigned ewg(long long total) { return (unsigned)((total + 255) / 256); }

extern "C" void kernel_launch(void* const* d_in, const int* in_sizes, int n_in,
                              void* d_out, int out_size, void* d_ws, size_t ws_size,
                              hipStream_t stream)
{
  (void)in_sizes; (void)n_in; (void)out_size; (void)ws_size;
  const float* x_in   = (const float*)d_in[0];
  const float* f_in   = (const float*)d_in[1];
  const float* t_ln1g = (const float*)d_in[3];
  const float* t_ln1b = (const float*)d_in[4];
  const float* t_wq   = (const float*)d_in[5];
  const float* t_bq   = (const float*)d_in[6];
  const float* t_wk   = (const float*)d_in[7];
  const float* t_bk   = (const float*)d_in[8];
  const float* t_wv   = (const float*)d_in[9];
  const float* t_bv   = (const float*)d_in[10];
  const float* t_wo   = (const float*)d_in[11];
  const float* t_bo   = (const float*)d_in[12];
  const float* t_ln2g = (const float*)d_in[13];
  const float* t_ln2b = (const float*)d_in[14];
  const float* t_fw1  = (const float*)d_in[15];
  const float* t_fb1  = (const float*)d_in[16];
  const float* t_fw2  = (const float*)d_in[17];
  const float* t_fb2  = (const float*)d_in[18];
  const float* s_rww  = (const float*)d_in[19];
  const float* s_rwb  = (const float*)d_in[20];
  const float* s_gkw  = (const float*)d_in[21];
  const float* s_gkb  = (const float*)d_in[22];
  const float* s_gvw  = (const float*)d_in[23];
  const float* s_gvb  = (const float*)d_in[24];
  const float* s_qng  = (const float*)d_in[25];
  const float* s_qnb  = (const float*)d_in[26];
  const float* s_kng  = (const float*)d_in[27];
  const float* s_knb  = (const float*)d_in[28];
  const float* s_vng  = (const float*)d_in[29];
  const float* s_vnb  = (const float*)d_in[30];
  const float* s_wq   = (const float*)d_in[31];
  const float* s_bq   = (const float*)d_in[32];
  const float* s_wk   = (const float*)d_in[33];
  const float* s_bk   = (const float*)d_in[34];
  const float* s_wv   = (const float*)d_in[35];
  const float* s_bv   = (const float*)d_in[36];
  const float* s_wo   = (const float*)d_in[37];
  const float* s_bo   = (const float*)d_in[38];
  const float* s_lng  = (const float*)d_in[39];
  const float* s_lnb  = (const float*)d_in[40];
  const float* s_fw1  = (const float*)d_in[41];
  const float* s_fb1  = (const float*)d_in[42];
  const float* s_fw2  = (const float*)d_in[43];
  const float* s_fb2  = (const float*)d_in[44];
  float* out = (float*)d_out;

  // ----- workspace arena (floats) -----
  float* ws = (float*)d_ws;
  size_t off = 0;
  auto alloc = [&](size_t n) { size_t o = off; off += (n + 255) & ~(size_t)255; return ws + o; };
  float* fS    = alloc(23040ULL * 512);
  float* fX    = alloc(23040ULL * 512);
  float* fT0   = alloc(25600ULL * 768);
  float* fT1   = alloc(25600ULL * 768);
  float* fT2   = alloc(25600ULL * 768);
  float* fT3   = alloc(25600ULL * 768);
  float* fT4   = alloc(25600ULL * 768);
  float* fT5   = alloc(25600ULL * 768);
  float* fT6   = alloc(25600ULL * 768);
  float* fBIG0 = alloc(23040ULL * 1960);
  float* fBIG1 = alloc(23040ULL * 1960);
  float* fSC   = alloc(256ULL * 400 * 400);
  float* fXF   = alloc(23040ULL * 768);
  float* fGATE = alloc(23040ULL * 256);
  float* fFN   = alloc(10ULL * 40 * 108 * 192);
  float* fGK   = alloc(10ULL * 144 * 768);
  float* fGV   = alloc(10ULL * 144 * 512);
  float* fKLNG = alloc(10ULL * 144 * 768);
  float* fVLNG = alloc(10ULL * 144 * 512);
  float* fKG   = alloc(10ULL * 144 * 512);
  float* fVG   = alloc(10ULL * 144 * 512);
  // transposed weights [N,K]
  float* tWQ  = alloc(512ULL * 512);
  float* tWK  = alloc(512ULL * 512);
  float* tWV  = alloc(512ULL * 512);
  float* tWO  = alloc(512ULL * 512);
  float* tFW1 = alloc(1960ULL * 512);
  float* tFW2 = alloc(512ULL * 1960);
  float* tRW  = alloc(256ULL * 768);
  float* tSWQ = alloc(512ULL * 768);
  float* tSWK = alloc(512ULL * 768);
  float* tSWV = alloc(512ULL * 512);
  float* tSWO = alloc(512ULL * 512);
  float* tSF1 = alloc(1960ULL * 512);
  float* tSF2 = alloc(512ULL * 1960);

  const float scl = 0.08838834764831845f;  // 1/sqrt(128)
  const long long Z = 0;

  // ----- weight transposes -----
  auto tr = [&](const float* src, float* dst, int K, int N) {
    long long tot = (long long)K * N;
    transpose_kernel<<<ewg(tot), 256, 0, stream>>>(src, dst, K, N, tot);
  };
  tr(t_wq, tWQ, 512, 512);   tr(t_wk, tWK, 512, 512);
  tr(t_wv, tWV, 512, 512);   tr(t_wo, tWO, 512, 512);
  tr(t_fw1, tFW1, 512, 1960); tr(t_fw2, tFW2, 1960, 512);
  tr(s_rww, tRW, 768, 256);
  tr(s_wq, tSWQ, 768, 512);  tr(s_wk, tSWK, 768, 512);
  tr(s_wv, tSWV, 512, 512);  tr(s_wo, tSWO, 512, 512);
  tr(s_fw1, tSF1, 512, 1960); tr(s_fw2, tSF2, 1960, 512);

  // ================= tmhsa =================
  ln_kernel<<<23040, 256, 0, stream>>>(x_in, fS, t_ln1g, t_ln1b, 512);
  gemm(stream, fS, tWQ, fT0, t_bq, nullptr, 23040, 512, 512, 512, 512, 512, 0,
       1, 1, Z, Z, Z, Z, Z, Z, Z, Z, 1.f);
  gemm(stream, fS, tWK, fT1, t_bk, nullptr, 23040, 512, 512, 512, 512, 512, 0,
       1, 1, Z, Z, Z, Z, Z, Z, Z, Z, 1.f);
  gemm(stream, fS, tWV, fT2, t_bv, nullptr, 23040, 512, 512, 512, 512, 512, 0,
       1, 1, Z, Z, Z, Z, Z, Z, Z, Z, 1.f);
  tmhsa_gather  <<<ewg(256LL * 400 * 128), 256, 0, stream>>>(fT0, t_bq, fT3);
  tmhsa_gather  <<<ewg(256LL * 400 * 128), 256, 0, stream>>>(fT1, t_bk, fT4);
  tmhsa_gather_t<<<ewg(256LL * 400 * 128), 256, 0, stream>>>(fT2, t_bv, fT5);
  gemm(stream, fT3, fT4, fSC, nullptr, nullptr, 400, 400, 128, 128, 128, 400, 0,
       256, 1, 400LL * 128, Z, 400LL * 128, Z, 400LL * 400, Z, Z, Z, scl);
  softmax_kernel<<<256 * 400, 256, 0, stream>>>(fSC, 400);
  gemm(stream, fSC, fT5, fT6, nullptr, nullptr, 400, 128, 400, 400, 400, 128, 0,
       256, 1, 400LL * 400, Z, 128LL * 400, Z, 400LL * 128, Z, Z, Z, 1.f);
  tmhsa_scatter<<<ewg(23040LL * 512), 256, 0, stream>>>(fT6, fT0);
  gemm(stream, fT0, tWO, fX, t_bo, x_in, 23040, 512, 512, 512, 512, 512, 512,
       1, 1, Z, Z, Z, Z, Z, Z, Z, Z, 1.f);

  // ================= fusion FFN 1 =================
  ln_kernel<<<23040, 256, 0, stream>>>(fX, fS, t_ln2g, t_ln2b, 512);
  gemm(stream, fS, tFW1, fBIG0, t_fb1, nullptr, 23040, 1960, 512, 512, 512, 1960, 0,
       1, 1, Z, Z, Z, Z, Z, Z, Z, Z, 1.f);
  foldnorm_kernel<<<ewg(10LL * 40 * 108 * 192), 256, 0, stream>>>(fBIG0, fFN);
  unfold_gelu_kernel<<<ewg(23040LL * 1960), 256, 0, stream>>>(fFN, fBIG1);
  gemm(stream, fBIG1, tFW2, fX, t_fb2, fX, 23040, 512, 1960, 1960, 1960, 512, 512,
       1, 1, Z, Z, Z, Z, Z, Z, Z, Z, 1.f);

  // ================= swmhsa =================
  concat_xf_kernel<<<ewg(23040LL * 768), 256, 0, stream>>>(fX, f_in, fXF);
  gemm(stream, fXF, tRW, fGATE, s_rwb, nullptr, 23040, 256, 768, 768, 768, 256, 0,
       1, 1, Z, Z, Z, Z, Z, Z, Z, Z, 1.f);
  gate_apply_kernel<<<ewg(23040LL * 256), 256, 0, stream>>>(f_in, fGATE, fXF);
  dwconv_kernel<<<ewg(10LL * 144 * 768), 256, 0, stream>>>(fXF, 768, s_gkw, s_gkb, fGK, 768);
  dwconv_kernel<<<ewg(10LL * 144 * 512), 256, 0, stream>>>(fX, 512, s_gvw, s_gvb, fGV, 512);
  win_gather_kernel<<<ewg(10LL * 40 * 64 * 768), 256, 0, stream>>>(fXF, 768, 768, fT0, 10LL * 40 * 64 * 768);
  win_gather_kernel<<<ewg(10LL * 40 * 64 * 512), 256, 0, stream>>>(fX, 512, 512, fT1, 10LL * 40 * 64 * 512);
  ln_kernel<<<25600, 256, 0, stream>>>(fT0, fT2, s_kng, s_knb, 768);   // K-norm window rows
  ln_kernel<<<1440, 256, 0, stream>>>(fGK, fKLNG, s_kng, s_knb, 768);  // K-norm global tokens
  ln_kernel<<<25600, 256, 0, stream>>>(fT0, fT0, s_qng, s_qnb, 768);   // Q-norm in place
  ln_kernel<<<25600, 256, 0, stream>>>(fT1, fT1, s_vng, s_vnb, 512);   // V-norm in place
  ln_kernel<<<1440, 256, 0, stream>>>(fGV, fVLNG, s_vng, s_vnb, 512);
  gemm(stream, fT0, tSWQ, fT3, s_bq, nullptr, 25600, 512, 768, 768, 768, 512, 0,
       1, 1, Z, Z, Z, Z, Z, Z, Z, Z, 1.f);                             // q
  gemm(stream, fT2, tSWK, fT4, s_bk, nullptr, 25600, 512, 768, 768, 768, 512, 0,
       1, 1, Z, Z, Z, Z, Z, Z, Z, Z, 1.f);                             // k (window)
  gemm(stream, fKLNG, tSWK, fKG, s_bk, nullptr, 1440, 512, 768, 768, 768, 512, 0,
       1, 1, Z, Z, Z, Z, Z, Z, Z, Z, 1.f);                             // k (global)
  gemm(stream, fT1, tSWV, fT5, s_bv, nullptr, 25600, 512, 512, 512, 512, 512, 0,
       1, 1, Z, Z, Z, Z, Z, Z, Z, Z, 1.f);                             // v (window)
  gemm(stream, fVLNG, tSWV, fVG, s_bv, nullptr, 1440, 512, 512, 512, 512, 512, 0,
       1, 1, Z, Z, Z, Z, Z, Z, Z, Z, 1.f);                             // v (global)
  kvcat_kernel  <<<ewg(10LL * 40 * 208 * 512), 256, 0, stream>>>(fT4, fKG, fBIG0);
  kvcat_t_kernel<<<ewg(10LL * 40 * 512 * 208), 256, 0, stream>>>(fT5, fVG, fBIG1);
  // scores: batch (bt*40+win, head); head slices via b2 stride
  gemm(stream, fT3, fBIG0, fSC, nullptr, nullptr, 64, 208, 128, 512, 512, 208, 0,
       1600, 4, 64LL * 512, 128LL, 208LL * 512, 128LL, 4LL * 64 * 208, 64LL * 208, Z, Z, scl);
  softmax_kernel<<<1600 * 64, 256, 0, stream>>>(fSC, 208);
  gemm(stream, fSC, fBIG1, fT6, nullptr, nullptr, 64, 128, 208, 208, 208, 512, 0,
       1600, 4, 4LL * 64 * 208, 64LL * 208, 512LL * 208, 128LL * 208, 64LL * 512, 128LL, Z, Z, 1.f);
  swin_scatter<<<ewg(23040LL * 512), 256, 0, stream>>>(fT6, fT2);
  gemm(stream, fT2, tSWO, fX, s_bo, fX, 23040, 512, 512, 512, 512, 512, 512,
       1, 1, Z, Z, Z, Z, Z, Z, Z, Z, 1.f);

  // ================= fusion FFN 2 (writes d_out) =================
  ln_kernel<<<23040, 256, 0, stream>>>(fX, fS, s_lng, s_lnb, 512);
  gemm(stream, fS, tSF1, fBIG0, s_fb1, nullptr, 23040, 1960, 512, 512, 512, 1960, 0,
       1, 1, Z, Z, Z, Z, Z, Z, Z, Z, 1.f);
  foldnorm_kernel<<<ewg(10LL * 40 * 108 * 192), 256, 0, stream>>>(fBIG0, fFN);
  unfold_gelu_kernel<<<ewg(23040LL * 1960), 256, 0, stream>>>(fFN, fBIG1);
  gemm(stream, fBIG1, tSF2, out, s_fb2, fX, 23040, 512, 1960, 1960, 1960, 512, 512,
       1, 1, Z, Z, Z, Z, Z, Z, Z, Z, 1.f);
}
